// RWKV_Tmix_x070_23012434771988
// MI455X (gfx1250) — matrix operand order repaired and checked by reading, not compiled, not run on hardware
//
#include <hip/hip_runtime.h>
#include <hip/hip_bf16.h>
#include <math.h>

// ---- problem constants ----
#define Bc    2
#define Tc    2048
#define Cdim  2048
#define Hc    32
#define HEADc 64
#define DWc   64
#define DAc   64
#define DVc   32
#define DGc   128
#define MROWS (Bc*Tc)                 // 4096
#define NELT  ((size_t)Bc*Tc*Cdim)   // 8388608
#define EPSGN 0.00064f

typedef __attribute__((ext_vector_type(16))) __bf16 bfx16;
typedef __attribute__((ext_vector_type(8)))  float  fx8;
typedef __attribute__((ext_vector_type(4)))  unsigned int u32x4_t;
typedef __attribute__((ext_vector_type(8)))  int          i32x8_t;
typedef __attribute__((ext_vector_type(4)))  int          i32x4_t;

#ifndef __has_builtin
#define __has_builtin(x) 0
#endif
#if __has_builtin(__builtin_amdgcn_tensor_load_to_lds) && __has_builtin(__builtin_amdgcn_s_wait_tensorcnt)
#define USE_TDM 1
#else
#define USE_TDM 0
#endif

__device__ __forceinline__ float sigm(float x) { return 1.0f / (1.0f + expf(-x)); }

// ------------------------------------------------------------------
// TDM: async 2D tile load Global -> LDS  (CDNA5 ISA)
// tile = rows x kelems (bf16), row stride = ld elements, packed into LDS.
// ------------------------------------------------------------------
#if USE_TDM
__device__ __forceinline__ void tdm_load_tile_2d(const __bf16* gsrc, unsigned lds_off,
                                                 unsigned kelems, unsigned rows, unsigned ld) {
  unsigned long long ga = (unsigned long long)(size_t)gsrc;
  // group0: [1:0]=count=1, [63:32]=lds_addr, [120:64]=global_addr, [127:126]=type=2
  u32x4_t g0 = { 1u,
                 lds_off,
                 (unsigned)(ga & 0xFFFFFFFFu),
                 (unsigned)((ga >> 32) & 0x01FFFFFFu) | 0x80000000u };
  // group1: data_size=1(2B); tensor_dim0=kelems; tensor_dim1=rows;
  //         tile_dim0=kelems; tile_dim1=rows; tensor_dim0_stride=ld
  i32x8_t g1;
  g1[0] = (int)(1u << 16);                                          // data_size, no masks/flags
  g1[1] = (int)((kelems & 0xFFFFu) << 16);                          // tensor_dim0[15:0] @ bits 63:48
  g1[2] = (int)(((kelems >> 16) & 0xFFFFu) | ((rows & 0xFFFFu) << 16)); // dim0 hi | dim1 lo
  g1[3] = (int)(((rows >> 16) & 0xFFFFu) | ((kelems & 0xFFFFu) << 16)); // dim1 hi | tile_dim0
  g1[4] = (int)(rows & 0xFFFFu);                                    // tile_dim1, tile_dim2=0
  g1[5] = (int)ld;                                                  // dim0_stride[31:0]
  g1[6] = 0;                                                        // dim0_stride hi | dim1_stride lo
  g1[7] = 0;
  i32x4_t g2 = { 0, 0, 0, 0 };
  i32x4_t g3 = { 0, 0, 0, 0 };
#if __clang_major__ >= 23
  i32x8_t g4 = { 0, 0, 0, 0, 0, 0, 0, 0 };
  __builtin_amdgcn_tensor_load_to_lds(g0, g1, g2, g3, g4, 0);
#else
  __builtin_amdgcn_tensor_load_to_lds(g0, g1, g2, g3, 0);
#endif
}
#endif

// ------------------------------------------------------------------
// fp32 -> bf16 conversions / activations
// ------------------------------------------------------------------
__global__ void cvt_bf16_kernel(const float* __restrict__ in, __bf16* __restrict__ out, int n) {
  int i = blockIdx.x * 256 + threadIdx.x;
  if (i < n) out[i] = (__bf16)in[i];
}
__global__ void tanh_bf16_kernel(const float* __restrict__ in, __bf16* __restrict__ out, int n) {
  int i = blockIdx.x * 256 + threadIdx.x;
  if (i < n) out[i] = (__bf16)tanhf(in[i]);
}
__global__ void sigmoid_bf16_kernel(const float* __restrict__ in, __bf16* __restrict__ out, int n) {
  int i = blockIdx.x * 256 + threadIdx.x;
  if (i < n) out[i] = (__bf16)sigm(in[i]);
}

// ------------------------------------------------------------------
// time-shift mixes: xq = x + (shift(x)-x) * x_q, write bf16
// ------------------------------------------------------------------
__global__ void prep_kernel(const float* __restrict__ x,
                            const float* __restrict__ cr, const float* __restrict__ cw,
                            const float* __restrict__ ck, const float* __restrict__ cv,
                            const float* __restrict__ ca, const float* __restrict__ cg,
                            __bf16* __restrict__ xr, __bf16* __restrict__ xw,
                            __bf16* __restrict__ xk, __bf16* __restrict__ xv,
                            __bf16* __restrict__ xa, __bf16* __restrict__ xg) {
  int idx = blockIdx.x * 256 + threadIdx.x;            // NELT multiple of 256
  int c   = idx & (Cdim - 1);
  int row = idx >> 11;
  int t   = row & (Tc - 1);
  float xc = x[idx];
  float xp = (t == 0) ? 0.0f : x[idx - Cdim];
  float xx = xp - xc;
  xr[idx] = (__bf16)(xc + xx * cr[c]);
  xw[idx] = (__bf16)(xc + xx * cw[c]);
  xk[idx] = (__bf16)(xc + xx * ck[c]);
  xv[idx] = (__bf16)(xc + xx * cv[c]);
  xa[idx] = (__bf16)(xc + xx * ca[c]);
  xg[idx] = (__bf16)(xc + xx * cg[c]);
}

// ------------------------------------------------------------------
// shared WMMA compute step: 2 M-tiles x 4 N-tiles, B tile from LDS (64x32 bf16)
// ------------------------------------------------------------------
__device__ __forceinline__ void wave_compute_step(const __bf16* __restrict__ xr0,
                                                  const __bf16* __restrict__ xr1,
                                                  const __bf16* bt, int k0,
                                                  int half, int mr, fx8 acc[2][4]) {
  union { bfx16 v; unsigned u[8]; } A0, A1;
#pragma unroll
  for (int e = 0; e < 8; ++e) {
    int kb = 2 * e + half * 16;
    A0.u[e] = *reinterpret_cast<const unsigned*>(xr0 + k0 + kb);
    A1.u[e] = *reinterpret_cast<const unsigned*>(xr1 + k0 + kb);
  }
#pragma unroll
  for (int t = 0; t < 4; ++t) {
    union { bfx16 v; u32x4_t q[2]; } Bm;
    const __bf16* row = bt + (t * 16 + mr) * 32 + half * 16;
    Bm.q[0] = *reinterpret_cast<const u32x4_t*>(row);
    Bm.q[1] = *reinterpret_cast<const u32x4_t*>(row + 8);
    acc[0][t] = __builtin_amdgcn_wmma_f32_16x16x32_bf16(false, A0.v, false, Bm.v,
                                                        (short)0, acc[0][t], false, false);
    acc[1][t] = __builtin_amdgcn_wmma_f32_16x16x32_bf16(false, A1.v, false, Bm.v,
                                                        (short)0, acc[1][t], false, false);
  }
}

// ------------------------------------------------------------------
// big-projection GEMM: O[M,N] = X[M,K] @ W[N,K]^T  (bf16 in, f32 out)
// block = 256 thr (8 waves), tile BM=256, BN=64, BK=32.
// B tile staged in LDS by the Tensor Data Mover, double-buffered.
// ------------------------------------------------------------------
__global__ __launch_bounds__(256) void gemm_bigproj_kernel(const __bf16* __restrict__ X,
                                                           const __bf16* __restrict__ W,
                                                           float* __restrict__ O,
                                                           int Mdim, int Ndim, int Kdim) {
  const int lane = threadIdx.x & 31;
  const int wave = threadIdx.x >> 5;
  const int half = lane >> 4;
  const int mr   = lane & 15;
  const int m0   = blockIdx.y * 256 + wave * 32;
  const int n0   = blockIdx.x * 64;

  __shared__ __align__(16) __bf16 smemB[2][64 * 32];

  const fx8 zero = {0.f, 0.f, 0.f, 0.f, 0.f, 0.f, 0.f, 0.f};
  fx8 acc[2][4];
#pragma unroll
  for (int i = 0; i < 2; ++i)
#pragma unroll
    for (int t = 0; t < 4; ++t) acc[i][t] = zero;

  const __bf16* xr0 = X + (size_t)(m0 + mr) * Kdim;
  const __bf16* xr1 = X + (size_t)(m0 + 16 + mr) * Kdim;
  const __bf16* wtile = W + (size_t)n0 * Kdim;
  const int KT = Kdim / 32;

#if USE_TDM
  const unsigned lds0 = (unsigned)(size_t)(&smemB[0][0]);
  const unsigned lds1 = (unsigned)(size_t)(&smemB[1][0]);
  if (wave == 0) tdm_load_tile_2d(wtile, lds0, 32, 64, (unsigned)Kdim);
  for (int it = 0; it < KT; ++it) {
    const int cur = it & 1;
    const int k0  = it * 32;
    if (wave == 0) {
      if (it + 1 < KT) {
        tdm_load_tile_2d(wtile + (k0 + 32), (cur ? lds0 : lds1), 32, 64, (unsigned)Kdim);
        __builtin_amdgcn_s_wait_tensorcnt(1);   // tile for this step has landed
      } else {
        __builtin_amdgcn_s_wait_tensorcnt(0);
      }
    }
    __syncthreads();
    if (k0 + 32 < Kdim) __builtin_prefetch(xr0 + k0 + 32, 0, 0);
    wave_compute_step(xr0, xr1, &smemB[cur][0], k0, half, mr, acc);
    __syncthreads();
  }
#else
  for (int it = 0; it < KT; ++it) {
    const int k0 = it * 32;
    __syncthreads();
    {
      const int trow = threadIdx.x >> 2, tpart = threadIdx.x & 3;   // 256 thr copy 4KB
      const u32x4_t* src =
          reinterpret_cast<const u32x4_t*>(wtile + (size_t)trow * Kdim + k0) + tpart;
      reinterpret_cast<u32x4_t*>(&smemB[0][trow * 32])[tpart] = *src;
    }
    __syncthreads();
    if (k0 + 32 < Kdim) __builtin_prefetch(xr0 + k0 + 32, 0, 0);
    wave_compute_step(xr0, xr1, &smemB[0][0], k0, half, mr, acc);
  }
#endif

#pragma unroll
  for (int mt = 0; mt < 2; ++mt)
#pragma unroll
    for (int t = 0; t < 4; ++t)
#pragma unroll
      for (int rI = 0; rI < 8; ++rI) {
        int mm = m0 + mt * 16 + half * 8 + rI;
        int nn = n0 + t * 16 + mr;
        O[(size_t)mm * Ndim + nn] = acc[mt][t][rI];
      }
}

// ------------------------------------------------------------------
// generic WMMA GEMM for the LoRA branches (small N or small K)
//   BT=false: W stored (K,N) row-major -> O = X @ W
// ------------------------------------------------------------------
template <int NTILES, bool BT>
__global__ __launch_bounds__(128) void gemm_bf16_kernel(const __bf16* __restrict__ X,
                                                        const __bf16* __restrict__ W,
                                                        float* __restrict__ O,
                                                        int Mdim, int Ndim, int Kdim) {
  const int lane = threadIdx.x & 31;
  const int wave = threadIdx.x >> 5;
  const int half = lane >> 4;
  const int mr   = lane & 15;
  const int m0   = (blockIdx.y * 4 + wave) * 16;
  const int n0   = blockIdx.x * (16 * NTILES);
  if (m0 >= Mdim) return;

  const fx8 zero = {0.f, 0.f, 0.f, 0.f, 0.f, 0.f, 0.f, 0.f};
  fx8 acc[NTILES];
#pragma unroll
  for (int t = 0; t < NTILES; ++t) acc[t] = zero;

  const __bf16* xrow = X + (size_t)(m0 + mr) * Kdim;

  for (int k0 = 0; k0 < Kdim; k0 += 32) {
    if (k0 + 32 < Kdim) __builtin_prefetch(xrow + k0 + 32, 0, 0);
    union { bfx16 v; unsigned u[8]; } A;
#pragma unroll
    for (int e = 0; e < 8; ++e) {
      int kb = 2 * e + half * 16;
      A.u[e] = *reinterpret_cast<const unsigned*>(xrow + k0 + kb);
    }
#pragma unroll
    for (int t = 0; t < NTILES; ++t) {
      const int n = n0 + t * 16 + mr;
      union { bfx16 v; unsigned u[8]; __bf16 h[16]; } Bm;
#pragma unroll
      for (int e = 0; e < 8; ++e) {
        int kb = 2 * e + half * 16;
        if (BT) {
          Bm.u[e] = *reinterpret_cast<const unsigned*>(W + (size_t)n * Kdim + k0 + kb);
        } else {
          Bm.h[2 * e]     = W[(size_t)(k0 + kb)     * Ndim + n];
          Bm.h[2 * e + 1] = W[(size_t)(k0 + kb + 1) * Ndim + n];
        }
      }
      acc[t] = __builtin_amdgcn_wmma_f32_16x16x32_bf16(false, A.v, false, Bm.v,
                                                       (short)0, acc[t], false, false);
    }
  }
#pragma unroll
  for (int t = 0; t < NTILES; ++t)
#pragma unroll
    for (int rI = 0; rI < 8; ++rI) {
      int mm = m0 + half * 8 + rI;
      int nn = n0 + t * 16 + mr;
      O[(size_t)mm * Ndim + nn] = acc[t][rI];
    }
}

// ------------------------------------------------------------------
// epilogues for the LoRA branches
// ------------------------------------------------------------------
__global__ void w_transform_kernel(float* __restrict__ w, const float* __restrict__ w0, int n) {
  int i = blockIdx.x * 256 + threadIdx.x;
  if (i >= n) return;
  int c = i & (Cdim - 1);
  float y  = w0[c] + w[i];
  float lw = -log1pf(expf(-y)) - 0.5f;
  w[i] = expf(-expf(lw));                // store decay directly
}
__global__ void a_transform_kernel(float* __restrict__ a, const float* __restrict__ a0, int n) {
  int i = blockIdx.x * 256 + threadIdx.x;
  if (i >= n) return;
  int c = i & (Cdim - 1);
  a[i] = sigm(a0[c] + a[i]);
}
__global__ void v_blend_kernel(float* __restrict__ v, const float* __restrict__ vl,
                               const float* __restrict__ v_first, const float* __restrict__ v0, int n) {
  int i = blockIdx.x * 256 + threadIdx.x;
  if (i >= n) return;
  int c = i & (Cdim - 1);
  float s = sigm(v0[c] + vl[i]);
  float vv = v[i];
  v[i] = vv + (v_first[i] - vv) * s;
}

// ------------------------------------------------------------------
// kk normalize per head, a=-kk, b=kk*a_sig, k *= (1+(a-1)*k_a)
// ------------------------------------------------------------------
__global__ void kkprep_kernel(float* __restrict__ k, const float* __restrict__ a,
                              const float* __restrict__ k_k, const float* __restrict__ k_a,
                              float* __restrict__ an, float* __restrict__ bv, int total) {
  int idx = blockIdx.x * 256 + threadIdx.x;
  if (idx >= total) return;
  int h = idx & (Hc - 1);
  size_t base = (size_t)idx * HEADc;
  const float* kkc = k_k + h * HEADc;
  const float* kac = k_a + h * HEADc;
  float ss = 0.f;
#pragma unroll 8
  for (int j = 0; j < HEADc; ++j) { float t = k[base + j] * kkc[j]; ss += t * t; }
  float inv = 1.0f / fmaxf(sqrtf(ss), 1e-12f);
#pragma unroll 8
  for (int j = 0; j < HEADc; ++j) {
    float kj  = k[base + j];
    float aj  = a[base + j];
    float kkn = kj * kkc[j] * inv;
    an[base + j] = -kkn;
    bv[base + j] = kkn * aj;
    k[base + j]  = kj * (1.0f + (aj - 1.0f) * kac[j]);
  }
}

// ------------------------------------------------------------------
// sequential RWKV-7 scan. grid = B*H, block = 256.
// ------------------------------------------------------------------
__global__ __launch_bounds__(256) void scan_kernel(const float* __restrict__ r,
                                                   const float* __restrict__ w,
                                                   const float* __restrict__ k,
                                                   const float* __restrict__ v,
                                                   const float* __restrict__ an,
                                                   const float* __restrict__ bv,
                                                   float* __restrict__ out) {
  const int bh  = blockIdx.x;
  const int b   = bh / Hc;
  const int h   = bh % Hc;
  const int tid = threadIdx.x;
  const int i   = tid >> 2;
  const int q   = tid & 3;

  __shared__ float sh[6 * HEADc];
  __shared__ float red[HEADc * 4];

  float S[16];
#pragma unroll
  for (int j = 0; j < 16; ++j) S[j] = 0.f;

  const size_t base = (size_t)b * Tc * Cdim + (size_t)h * HEADc;

  for (int t = 0; t < Tc; ++t) {
    const size_t off = base + (size_t)t * Cdim;
    for (int s = tid; s < 6 * HEADc; s += 256) {
      int arr = s >> 6, j = s & 63;
      const float* p = (arr == 0) ? r : (arr == 1) ? w : (arr == 2) ? k
                     : (arr == 3) ? v : (arr == 4) ? an : bv;
      sh[s] = p[off + j];
    }
    __syncthreads();
    const float* rs = sh;           const float* ws = sh + 64;
    const float* ks = sh + 128;     const float* vs = sh + 192;
    const float* as = sh + 256;     const float* bs = sh + 320;

    float pa = 0.f;
#pragma unroll
    for (int j = 0; j < 16; ++j) pa += S[j] * as[q * 16 + j];
    red[i * 4 + q] = pa;
    __syncthreads();
    float sa = red[i * 4] + red[i * 4 + 1] + red[i * 4 + 2] + red[i * 4 + 3];
    __syncthreads();

    float vi = vs[i];
    float po = 0.f;
#pragma unroll
    for (int j = 0; j < 16; ++j) {
      int jj = q * 16 + j;
      float s = S[j] * ws[jj] + sa * bs[jj] + vi * ks[jj];
      S[j] = s;
      po += s * rs[jj];
    }
    red[i * 4 + q] = po;
    __syncthreads();
    if (q == 0)
      out[off + i] = red[i * 4] + red[i * 4 + 1] + red[i * 4 + 2] + red[i * 4 + 3];
    __syncthreads();
  }
}

// ------------------------------------------------------------------
// per-head GroupNorm + rk*v residual + gating -> bf16
// ------------------------------------------------------------------
__global__ __launch_bounds__(64) void gn_gate_kernel(const float* __restrict__ ov,
                                                     const float* __restrict__ r,
                                                     const float* __restrict__ k,
                                                     const float* __restrict__ v,
                                                     const float* __restrict__ g,
                                                     const float* __restrict__ r_k,
                                                     const float* __restrict__ ln_g,
                                                     const float* __restrict__ ln_b,
                                                     __bf16* __restrict__ xog) {
  const int blk = blockIdx.x;
  const int h   = blk & (Hc - 1);
  const int tid = threadIdx.x;
  const size_t base = (size_t)blk * HEADc;
  const int c = h * HEADc + tid;

  __shared__ float sh[HEADc];
  float val = ov[base + tid];

  sh[tid] = val; __syncthreads();
  for (int s = 32; s > 0; s >>= 1) { if (tid < s) sh[tid] += sh[tid + s]; __syncthreads(); }
  float mean = sh[0] * (1.0f / HEADc);
  __syncthreads();

  float d = val - mean;
  sh[tid] = d * d; __syncthreads();
  for (int s = 32; s > 0; s >>= 1) { if (tid < s) sh[tid] += sh[tid + s]; __syncthreads(); }
  float var = sh[0] * (1.0f / HEADc);
  __syncthreads();

  float xn = d * rsqrtf(var + EPSGN) * ln_g[c] + ln_b[c];

  sh[tid] = r[base + tid] * k[base + tid] * r_k[c]; __syncthreads();
  for (int s = 32; s > 0; s >>= 1) { if (tid < s) sh[tid] += sh[tid + s]; __syncthreads(); }
  float rk = sh[0];
  __syncthreads();

  float xo = xn + rk * v[base + tid];
  xog[base + tid] = (__bf16)(xo * g[base + tid]);
}

// ------------------------------------------------------------------
// launcher
// ------------------------------------------------------------------
extern "C" void kernel_launch(void* const* d_in, const int* in_sizes, int n_in,
                              void* d_out, int out_size, void* d_ws, size_t ws_size,
                              hipStream_t stream) {
  const float* x       = (const float*)d_in[0];
  const float* v_first = (const float*)d_in[1];
  const float* cr      = (const float*)d_in[2];
  const float* cw      = (const float*)d_in[3];
  const float* ck      = (const float*)d_in[4];
  const float* cv      = (const float*)d_in[5];
  const float* ca      = (const float*)d_in[6];
  const float* cg      = (const float*)d_in[7];
  const float* w0      = (const float*)d_in[8];
  const float* w1      = (const float*)d_in[9];
  const float* w2      = (const float*)d_in[10];
  const float* a0      = (const float*)d_in[11];
  const float* a1      = (const float*)d_in[12];
  const float* a2      = (const float*)d_in[13];
  const float* v0      = (const float*)d_in[14];
  const float* v1      = (const float*)d_in[15];
  const float* v2      = (const float*)d_in[16];
  const float* g1      = (const float*)d_in[17];
  const float* g2      = (const float*)d_in[18];
  const float* k_k     = (const float*)d_in[19];
  const float* k_a     = (const float*)d_in[20];
  const float* r_k     = (const float*)d_in[21];
  const float* W_r     = (const float*)d_in[22];
  const float* W_k     = (const float*)d_in[23];
  const float* W_v     = (const float*)d_in[24];
  const float* W_o     = (const float*)d_in[25];
  const float* ln_g    = (const float*)d_in[26];
  const float* ln_b    = (const float*)d_in[27];
  float* y = (float*)d_out;

  // ---- workspace arena ----
  char* ws = (char*)d_ws;
  size_t cur = 0;
  auto arena = [&](size_t bytes) -> void* {
    void* p = ws + cur;
    cur = (cur + bytes + 255) & ~(size_t)255;
    return p;
  };
  const size_t f32big = NELT * sizeof(float);
  const size_t bf16big = NELT * sizeof(__bf16);

  float* r_f  = (float*)arena(f32big);
  float* w_f  = (float*)arena(f32big);
  float* k_f  = (float*)arena(f32big);
  float* v_f  = (float*)arena(f32big);
  float* vl_f = (float*)arena(f32big);
  float* a_f  = (float*)arena(f32big);
  float* g_f  = (float*)arena(f32big);
  float* an_f = (float*)arena(f32big);
  float* bv_f = (float*)arena(f32big);
  float* o_f  = (float*)arena(f32big);

  __bf16* xr_h  = (__bf16*)arena(bf16big);
  __bf16* xw_h  = (__bf16*)arena(bf16big);
  __bf16* xk_h  = (__bf16*)arena(bf16big);
  __bf16* xv_h  = (__bf16*)arena(bf16big);
  __bf16* xa_h  = (__bf16*)arena(bf16big);
  __bf16* xg_h  = (__bf16*)arena(bf16big);
  __bf16* xog_h = (__bf16*)arena(bf16big);

  const size_t wcc = (size_t)Cdim * Cdim;
  __bf16* Wr_h = (__bf16*)arena(wcc * 2);
  __bf16* Wk_h = (__bf16*)arena(wcc * 2);
  __bf16* Wv_h = (__bf16*)arena(wcc * 2);
  __bf16* Wo_h = (__bf16*)arena(wcc * 2);
  __bf16* w1_h = (__bf16*)arena((size_t)Cdim * DWc * 2);
  __bf16* w2_h = (__bf16*)arena((size_t)DWc * Cdim * 2);
  __bf16* a1_h = (__bf16*)arena((size_t)Cdim * DAc * 2);
  __bf16* a2_h = (__bf16*)arena((size_t)DAc * Cdim * 2);
  __bf16* v1_h = (__bf16*)arena((size_t)Cdim * DVc * 2);
  __bf16* v2_h = (__bf16*)arena((size_t)DVc * Cdim * 2);
  __bf16* g1_h = (__bf16*)arena((size_t)Cdim * DGc * 2);
  __bf16* g2_h = (__bf16*)arena((size_t)DGc * Cdim * 2);

  float*  hw_f = (float*)arena((size_t)MROWS * DWc * 4);
  float*  ha_f = (float*)arena((size_t)MROWS * DAc * 4);
  float*  hv_f = (float*)arena((size_t)MROWS * DVc * 4);
  float*  hg_f = (float*)arena((size_t)MROWS * DGc * 4);
  __bf16* hw_h = (__bf16*)arena((size_t)MROWS * DWc * 2);
  __bf16* ha_h = (__bf16*)arena((size_t)MROWS * DAc * 2);
  __bf16* hv_h = (__bf16*)arena((size_t)MROWS * DVc * 2);
  __bf16* hg_h = (__bf16*)arena((size_t)MROWS * DGc * 2);

  auto cvt = [&](const float* in, __bf16* out, int n) {
    cvt_bf16_kernel<<<(n + 255) / 256, 256, 0, stream>>>(in, out, n);
  };

  // 1. weight conversions
  cvt(W_r, Wr_h, (int)wcc); cvt(W_k, Wk_h, (int)wcc);
  cvt(W_v, Wv_h, (int)wcc); cvt(W_o, Wo_h, (int)wcc);
  cvt(w1, w1_h, Cdim * DWc); cvt(w2, w2_h, DWc * Cdim);
  cvt(a1, a1_h, Cdim * DAc); cvt(a2, a2_h, DAc * Cdim);
  cvt(v1, v1_h, Cdim * DVc); cvt(v2, v2_h, DVc * Cdim);
  cvt(g1, g1_h, Cdim * DGc); cvt(g2, g2_h, DGc * Cdim);

  // 2. time-shift mixes
  prep_kernel<<<(int)(NELT / 256), 256, 0, stream>>>(x, cr, cw, ck, cv, ca, cg,
                                                     xr_h, xw_h, xk_h, xv_h, xa_h, xg_h);

  const dim3 blkB(256);
  const dim3 gridBig(Cdim / 64, MROWS / 256);   // (32, 16)
  const dim3 blkL(128);
  const dim3 gridCC(Cdim / 64, MROWS / 64);

  // 3. big projections (TDM-staged WMMA GEMM)
  gemm_bigproj_kernel<<<gridBig, blkB, 0, stream>>>(xr_h, Wr_h, r_f, MROWS, Cdim, Cdim);
  gemm_bigproj_kernel<<<gridBig, blkB, 0, stream>>>(xk_h, Wk_h, k_f, MROWS, Cdim, Cdim);
  gemm_bigproj_kernel<<<gridBig, blkB, 0, stream>>>(xv_h, Wv_h, v_f, MROWS, Cdim, Cdim);

  // 4. LoRA stage 1 (O = X @ W, W stored (K,N))
  gemm_bf16_kernel<4, false><<<dim3(DWc / 64, MROWS / 64), blkL, 0, stream>>>(xw_h, w1_h, hw_f, MROWS, DWc, Cdim);
  gemm_bf16_kernel<4, false><<<dim3(DAc / 64, MROWS / 64), blkL, 0, stream>>>(xa_h, a1_h, ha_f, MROWS, DAc, Cdim);
  gemm_bf16_kernel<2, false><<<dim3(DVc / 32, MROWS / 64), blkL, 0, stream>>>(xv_h, v1_h, hv_f, MROWS, DVc, Cdim);
  gemm_bf16_kernel<4, false><<<dim3(DGc / 64, MROWS / 64), blkL, 0, stream>>>(xg_h, g1_h, hg_f, MROWS, DGc, Cdim);

  // 5. inter-stage activations
  tanh_bf16_kernel<<<(MROWS * DWc + 255) / 256, 256, 0, stream>>>(hw_f, hw_h, MROWS * DWc);
  cvt(ha_f, ha_h, MROWS * DAc);
  cvt(hv_f, hv_h, MROWS * DVc);
  sigmoid_bf16_kernel<<<(MROWS * DGc + 255) / 256, 256, 0, stream>>>(hg_f, hg_h, MROWS * DGc);

  // 6. LoRA stage 2 (K small)
  gemm_bf16_kernel<4, false><<<gridCC, blkL, 0, stream>>>(hw_h, w2_h, w_f,  MROWS, Cdim, DWc);
  gemm_bf16_kernel<4, false><<<gridCC, blkL, 0, stream>>>(ha_h, a2_h, a_f,  MROWS, Cdim, DAc);
  gemm_bf16_kernel<4, false><<<gridCC, blkL, 0, stream>>>(hv_h, v2_h, vl_f, MROWS, Cdim, DVc);
  gemm_bf16_kernel<4, false><<<gridCC, blkL, 0, stream>>>(hg_h, g2_h, g_f,  MROWS, Cdim, DGc);

  // 7. scalar transforms
  const int nel = (int)NELT;
  w_transform_kernel<<<nel / 256, 256, 0, stream>>>(w_f, w0, nel);
  a_transform_kernel<<<nel / 256, 256, 0, stream>>>(a_f, a0, nel);
  v_blend_kernel<<<nel / 256, 256, 0, stream>>>(v_f, vl_f, v_first, v0, nel);

  // 8. kk normalize + k update + a/b vectors
  const int bth = Bc * Tc * Hc;
  kkprep_kernel<<<(bth + 255) / 256, 256, 0, stream>>>(k_f, a_f, k_k, k_a, an_f, bv_f, bth);

  // 9. sequential scan
  scan_kernel<<<Bc * Hc, 256, 0, stream>>>(r_f, w_f, k_f, v_f, an_f, bv_f, o_f);

  // 10. groupnorm + rk*v + gating
  gn_gate_kernel<<<bth, HEADc, 0, stream>>>(o_f, r_f, k_f, v_f, g_f, r_k, ln_g, ln_b, xog_h);

  // 11. output projection straight into d_out
  gemm_bigproj_kernel<<<gridBig, blkB, 0, stream>>>(xog_h, Wo_h, y, MROWS, Cdim, Cdim);

  // 12. second tuple element: v_first passthrough
  hipMemcpyAsync(y + NELT, v_first, NELT * sizeof(float), hipMemcpyDeviceToDevice, stream);
}
